// Attention_55293408968939
// MI455X (gfx1250) — matrix operand order repaired and checked by reading, not compiled, not run on hardware
//
#include <hip/hip_runtime.h>
#include <hip/hip_bf16.h>

// ---------------------------------------------------------------------------
// ViT attention block on gfx1250 (MI455X), bf16 WMMA + fp32 accumulate.
//   K1: fp32 -> bf16 conversion of x, qkv_w, proj_w
//   K2: QKV GEMM (128x128 block tile, 2Mx4N wave tile, async LDS staging)
//       + bias + RMSNorm + 3D RoPE -> q/k bf16, v^T bf16
//   K3: attention per (head, 16-query tile): S=QK^T (WMMA, hoisted Q frags),
//       fp32 softmax in LDS (128KB strip), O=PV (WMMA) -> bf16
//   K4: projection GEMM (same blocked GEMM) + bias -> fp32 output
// Whole problem (~100 MB) fits in the 192 MB L2; ~103 GFLOP total => compute
// bound, so everything heavy runs on v_wmma_f32_16x16x32_bf16.
// ---------------------------------------------------------------------------

#define NTOK 2048
#define CDIM 2048
#define NHEAD 16
#define HD 128
#define KCHUNK 256          // K-chunk staged in LDS per GEMM iteration
#define KPAD 272            // padded row stride (halves): 544B -> 2-way banks max

typedef __attribute__((ext_vector_type(16))) __bf16 v16bf;
typedef __attribute__((ext_vector_type(8)))  float  v8f;
typedef __attribute__((ext_vector_type(8)))  unsigned short u16x8;
typedef __attribute__((ext_vector_type(4)))  float  f32x4;

union FragBF {
    v16bf v;
    u16x8 h[2];
    unsigned short u[16];
};

__device__ __forceinline__ unsigned short f2bf_bits(float f) {
    unsigned u = __float_as_uint(f);
    return (unsigned short)((u + 0x7FFFu + ((u >> 16) & 1u)) >> 16);  // RNE
}

__device__ __forceinline__ v8f wmma_bf16(const FragBF& a, const FragBF& b, v8f c) {
    return __builtin_amdgcn_wmma_f32_16x16x32_bf16(false, a.v, false, b.v,
                                                   (short)0, c, false, false);
}

// CDNA5 async global->LDS copy (ASYNCcnt path, CDNA5 ISA).
// LDS byte offset = low 32 bits of the generic pointer (aperture rule §10.2).
__device__ __forceinline__ void async_load_lds16(void* lds_dst, const void* gsrc) {
    unsigned loff = (unsigned)(size_t)lds_dst;
    unsigned long long ga = (unsigned long long)(size_t)gsrc;
    asm volatile("global_load_async_to_lds_b128 %0, %1, off"
                 :: "v"(loff), "v"(ga) : "memory");
}
__device__ __forceinline__ void wait_async_zero() {
    asm volatile("s_wait_asynccnt 0" ::: "memory");
}

// Stage a 128-row x KCHUNK A panel (bf16) into padded LDS via async copies.
__device__ __forceinline__ void stage_a_chunk(unsigned short* As,
                                              const unsigned short* gA,
                                              int n0, int chunk, int tid) {
    // 128 rows * 32 vec8-chunks = 4096; 256 threads -> 16 asyncs each
    for (int i = tid; i < 128 * (KCHUNK / 8); i += 256) {
        const int r = i >> 5, c8 = i & 31;
        async_load_lds16(As + r * KPAD + c8 * 8,
                         gA + ((size_t)(n0 + r)) * CDIM + chunk + c8 * 8);
    }
    wait_async_zero();
}

// ---------------------------------------------------------------------------
// K1: fp32 -> bf16 convert (grid-stride)
// ---------------------------------------------------------------------------
__global__ void cvt_f32_bf16_kernel(const float* __restrict__ src,
                                    unsigned short* __restrict__ dst, int n) {
    int i = blockIdx.x * blockDim.x + threadIdx.x;
    int stride = gridDim.x * blockDim.x;
    for (; i < n; i += stride) dst[i] = f2bf_bits(src[i]);
}

// ---------------------------------------------------------------------------
// K2: QKV GEMM fused with bias + RMSNorm + RoPE.
// Block = 256 threads (8 waves), block tile 128 rows x 128 cols (one head of
// q/k/v). Wave tile = 2M x 4N (8 WMMAs per k-step, A/B fragment reuse).
// grid = (NTOK/128, 3*NHEAD).
// ---------------------------------------------------------------------------
__global__ void qkv_gemm_fused_kernel(const unsigned short* __restrict__ xb,
                                      const unsigned short* __restrict__ wq,
                                      const float* __restrict__ bias,
                                      const float* __restrict__ qnw,
                                      const float* __restrict__ knw,
                                      unsigned short* __restrict__ qout,
                                      unsigned short* __restrict__ kout,
                                      unsigned short* __restrict__ vtout) {
    extern __shared__ char smem[];
    unsigned short* As = (unsigned short*)smem;            // 128*KPAD bf16 = 68KB
    float* Cs  = (float*)smem;                              // aliased after GEMM: 64KB
    float* red = (float*)(smem + 128 * KPAD * 2);           // 256 f32

    const int n0      = blockIdx.x * 128;
    const int gy      = blockIdx.y;        // 0..47
    const int sIdx    = gy >> 4;           // 0=q 1=k 2=v
    const int head    = gy & 15;
    const int colbase = gy * HD;
    const int tid  = threadIdx.x;
    const int wave = tid >> 5;
    const int lane = tid & 31;
    const int lm   = lane & 15;
    const int hi   = lane >> 4;
    const int mt0  = (wave & 3) * 2;       // 2 M tiles per wave
    const int nb0  = (wave >> 2) * 64;     // 4 N tiles per wave

    v8f acc[2][4] = {};
    for (int chunk = 0; chunk < CDIM; chunk += KCHUNK) {
        __syncthreads();
        stage_a_chunk(As, xb, n0, chunk, tid);
        __syncthreads();
#pragma unroll
        for (int ks = 0; ks < KCHUNK / 32; ++ks) {
            const int kl = ks * 32;
            FragBF a0, a1;
            const u16x8* ar0 = (const u16x8*)(As + (mt0 * 16 + lm) * KPAD + kl + hi * 8);
            const u16x8* ar1 = (const u16x8*)(As + (mt0 * 16 + 16 + lm) * KPAD + kl + hi * 8);
            a0.h[0] = ar0[0]; a0.h[1] = ar0[2];
            a1.h[0] = ar1[0]; a1.h[1] = ar1[2];
#pragma unroll
            for (int nt = 0; nt < 4; ++nt) {
                FragBF b;
                const unsigned short* brow =
                    wq + (size_t)(colbase + nb0 + nt * 16 + lm) * CDIM + chunk + kl + hi * 8;
                b.h[0] = ((const u16x8*)brow)[0];
                b.h[1] = ((const u16x8*)brow)[2];
                acc[0][nt] = wmma_bf16(a0, b, acc[0][nt]);
                acc[1][nt] = wmma_bf16(a1, b, acc[1][nt]);
            }
        }
    }
    __syncthreads();   // all LDS A reads done; Cs aliases As

    // C tile + bias -> LDS staging (fp32, 128 x 128)
#pragma unroll
    for (int mt = 0; mt < 2; ++mt)
#pragma unroll
        for (int nt = 0; nt < 4; ++nt) {
            const int c = nb0 + nt * 16 + lm;
            const float bval = bias[colbase + c];
#pragma unroll
            for (int j = 0; j < 8; ++j) {
                const int m = (mt0 + mt) * 16 + j + hi * 8;
                Cs[m * HD + c] = acc[mt][nt][j] + bval;
            }
        }
    __syncthreads();

    if (sIdx == 2) {
        // V: store transposed bf16 vt[head][d][n] (keys contiguous for P@V B-frags)
        const int d = tid >> 1;
        const int half = tid & 1;
        unsigned short tmp[64];
#pragma unroll
        for (int i = 0; i < 64; ++i)
            tmp[i] = f2bf_bits(Cs[(half * 64 + i) * HD + d]);
        u16x8* dstv = (u16x8*)(vtout + ((size_t)head * HD + d) * NTOK + n0 + half * 64);
#pragma unroll
        for (int i = 0; i < 8; ++i) dstv[i] = ((const u16x8*)tmp)[i];
    } else {
        // Q/K: RMSNorm (fp32) + 3D RoPE, bf16 store [head][n][d]
        const float* nw = (sIdx == 0) ? qnw : knw;
        unsigned short* outp = (sIdx == 0) ? qout : kout;
        const int r = tid >> 1;       // row 0..127
        const int seg = tid & 1;      // 64 elems each
        float* row = Cs + r * HD;

        float ss = 0.f;
        for (int i = 0; i < 64; ++i) { float v = row[seg * 64 + i]; ss += v * v; }
        red[r * 2 + seg] = ss;
        __syncthreads();
        const float rstd = rsqrtf((red[r * 2] + red[r * 2 + 1]) * (1.0f / 128.0f) + 1e-6f);

        // dt=44 (22 pairs), dh=42 (21), dw=42 (21); token n = t*256 + h*16 + w
        const int n = n0 + r;
        const int tpos = n >> 8, hpos = (n >> 4) & 15, wpos = n & 15;
        unsigned short ob[64];
        for (int pi = 0; pi < 32; ++pi) {
            const int p = seg * 32 + pi;
            float pos, dnm; int pb;
            if (p < 22)      { pos = (float)tpos; pb = p;      dnm = 44.f; }
            else if (p < 43) { pos = (float)hpos; pb = p - 22; dnm = 42.f; }
            else             { pos = (float)wpos; pb = p - 43; dnm = 42.f; }
            const float ang = pos * __powf(10000.0f, -2.0f * (float)pb / dnm);
            float s, c;
            __sincosf(ang, &s, &c);
            const float xr = row[2 * p] * rstd * nw[2 * p];
            const float xi = row[2 * p + 1] * rstd * nw[2 * p + 1];
            ob[2 * pi]     = f2bf_bits(xr * c - xi * s);
            ob[2 * pi + 1] = f2bf_bits(xr * s + xi * c);
        }
        u16x8* dst = (u16x8*)(outp + ((size_t)head * NTOK + n) * HD + seg * 64);
#pragma unroll
        for (int i = 0; i < 8; ++i) dst[i] = ((const u16x8*)ob)[i];
    }
}

// ---------------------------------------------------------------------------
// K3: attention.  grid = (NTOK/16, NHEAD), block = 128 threads (4 waves).
// Q fragments hoisted to registers (invariant over key tiles). Full 16x2048
// fp32 score strip lives in LDS (128KB fits the 320KB WGP LDS) -> plain
// softmax, no flash bookkeeping.
// ---------------------------------------------------------------------------
__global__ void attn_kernel(const unsigned short* __restrict__ qb,
                            const unsigned short* __restrict__ kb_,
                            const unsigned short* __restrict__ vt,
                            unsigned short* __restrict__ aout) {
    extern __shared__ char smem[];
    float* S   = (float*)smem;                        // 16*2048 f32 = 128KB
    float* red = (float*)(smem + 16 * NTOK * 4);      // 2*128 f32

    const int n0   = blockIdx.x * 16;
    const int head = blockIdx.y;
    const int tid  = threadIdx.x;   // 128
    const int wave = tid >> 5;
    const int lane = tid & 31;
    const int lm   = lane & 15;
    const int hi   = lane >> 4;
    const float scale = 0.08838834764831845f;  // 128^-0.5

    // Hoist Q A-fragments (4 k-steps over D=128) straight from global.
    FragBF qf[4];
#pragma unroll
    for (int ks = 0; ks < 4; ++ks) {
        const unsigned short* qrow =
            qb + ((size_t)head * NTOK + n0 + lm) * HD + ks * 32 + hi * 8;
        qf[ks].h[0] = ((const u16x8*)qrow)[0];
        qf[ks].h[1] = ((const u16x8*)qrow)[2];   // +16 halves
    }

    // ---- S = Q @ K^T ---- wave handles keys [wave*512, +512)
    const unsigned short* kbase = kb_ + (size_t)head * NTOK * HD;
    for (int kt = 0; kt < 32; ++kt) {
        const int key0 = wave * 512 + kt * 16;
        v8f acc = {};
#pragma unroll
        for (int ks = 0; ks < 4; ++ks) {
            FragBF b;
            const unsigned short* brow =
                kbase + (size_t)(key0 + lm) * HD + ks * 32 + hi * 8;
            b.h[0] = ((const u16x8*)brow)[0];
            b.h[1] = ((const u16x8*)brow)[2];
            acc = wmma_bf16(qf[ks], b, acc);
        }
#pragma unroll
        for (int j = 0; j < 8; ++j)
            S[(j + hi * 8) * NTOK + key0 + lm] = acc[j];
    }
    __syncthreads();

    // ---- softmax (fp32 in LDS): 8 threads per row, 256 elems each
    {
        const int r = tid >> 3, seg = tid & 7;
        float* row = S + r * NTOK;
        const int base = seg * 256;
        float m = -3.0e38f;
        for (int i = 0; i < 256; ++i) m = fmaxf(m, row[base + i]);
        red[r * 8 + seg] = m;
        __syncthreads();
        float rm = red[r * 8];
#pragma unroll
        for (int i = 1; i < 8; ++i) rm = fmaxf(rm, red[r * 8 + i]);
        float lsum = 0.f;
        for (int i = 0; i < 256; ++i) {
            const float p = __expf((row[base + i] - rm) * scale);
            row[base + i] = p;
            lsum += p;
        }
        red[128 + r * 8 + seg] = lsum;
        __syncthreads();
        float tot = 0.f;
#pragma unroll
        for (int i = 0; i < 8; ++i) tot += red[128 + r * 8 + i];
        const float inv = 1.0f / tot;
        for (int i = 0; i < 256; ++i) row[base + i] *= inv;
    }
    __syncthreads();

    // ---- O = P @ V ---- wave owns d-tiles {2w, 2w+1}; A frag shared
    const unsigned short* vbase = vt + (size_t)head * HD * NTOK;
    v8f oacc0 = {}, oacc1 = {};
    for (int kb2 = 0; kb2 < NTOK; kb2 += 32) {
        FragBF a;
        const float* srow = S + lm * NTOK + kb2 + hi * 16;
        f32x4 a0 = ((const f32x4*)srow)[0];
        f32x4 a1 = ((const f32x4*)srow)[1];
        f32x4 a2 = ((const f32x4*)(srow + 8))[0];
        f32x4 a3 = ((const f32x4*)(srow + 8))[1];
#pragma unroll
        for (int i = 0; i < 4; ++i) {
            a.u[i]      = f2bf_bits(a0[i]);
            a.u[4 + i]  = f2bf_bits(a1[i]);
            a.u[8 + i]  = f2bf_bits(a2[i]);
            a.u[12 + i] = f2bf_bits(a3[i]);
        }
        FragBF b0, b1;
        const unsigned short* br0 =
            vbase + (size_t)((wave * 2 + 0) * 16 + lm) * NTOK + kb2 + hi * 16;
        const unsigned short* br1 =
            vbase + (size_t)((wave * 2 + 1) * 16 + lm) * NTOK + kb2 + hi * 16;
        b0.h[0] = ((const u16x8*)br0)[0]; b0.h[1] = ((const u16x8*)br0)[1];
        b1.h[0] = ((const u16x8*)br1)[0]; b1.h[1] = ((const u16x8*)br1)[1];
        oacc0 = wmma_bf16(a, b0, oacc0);
        oacc1 = wmma_bf16(a, b1, oacc1);
    }
#pragma unroll
    for (int j = 0; j < 8; ++j) {
        const int m = j + hi * 8;
        aout[(size_t)(n0 + m) * CDIM + head * HD + (wave * 2 + 0) * 16 + lm] = f2bf_bits(oacc0[j]);
        aout[(size_t)(n0 + m) * CDIM + head * HD + (wave * 2 + 1) * 16 + lm] = f2bf_bits(oacc1[j]);
    }
}

// ---------------------------------------------------------------------------
// K4: projection GEMM + bias -> fp32. Same blocked GEMM as K2, plain epilogue.
// grid = (NTOK/128, CDIM/128), block = 256 threads.
// ---------------------------------------------------------------------------
__global__ void proj_gemm_kernel(const unsigned short* __restrict__ ab,
                                 const unsigned short* __restrict__ wp,
                                 const float* __restrict__ bias,
                                 float* __restrict__ out) {
    extern __shared__ char smem[];
    unsigned short* As = (unsigned short*)smem;   // 128*KPAD bf16

    const int n0      = blockIdx.x * 128;
    const int colbase = blockIdx.y * 128;
    const int tid  = threadIdx.x;
    const int wave = tid >> 5;
    const int lane = tid & 31;
    const int lm   = lane & 15;
    const int hi   = lane >> 4;
    const int mt0  = (wave & 3) * 2;
    const int nb0  = (wave >> 2) * 64;

    v8f acc[2][4] = {};
    for (int chunk = 0; chunk < CDIM; chunk += KCHUNK) {
        __syncthreads();
        stage_a_chunk(As, ab, n0, chunk, tid);
        __syncthreads();
#pragma unroll
        for (int ks = 0; ks < KCHUNK / 32; ++ks) {
            const int kl = ks * 32;
            FragBF a0, a1;
            const u16x8* ar0 = (const u16x8*)(As + (mt0 * 16 + lm) * KPAD + kl + hi * 8);
            const u16x8* ar1 = (const u16x8*)(As + (mt0 * 16 + 16 + lm) * KPAD + kl + hi * 8);
            a0.h[0] = ar0[0]; a0.h[1] = ar0[2];
            a1.h[0] = ar1[0]; a1.h[1] = ar1[2];
#pragma unroll
            for (int nt = 0; nt < 4; ++nt) {
                FragBF b;
                const unsigned short* brow =
                    wp + (size_t)(colbase + nb0 + nt * 16 + lm) * CDIM + chunk + kl + hi * 8;
                b.h[0] = ((const u16x8*)brow)[0];
                b.h[1] = ((const u16x8*)brow)[2];
                acc[0][nt] = wmma_bf16(a0, b, acc[0][nt]);
                acc[1][nt] = wmma_bf16(a1, b, acc[1][nt]);
            }
        }
    }

#pragma unroll
    for (int mt = 0; mt < 2; ++mt)
#pragma unroll
        for (int nt = 0; nt < 4; ++nt) {
            const int c = colbase + nb0 + nt * 16 + lm;
            const float bval = bias[c];
#pragma unroll
            for (int j = 0; j < 8; ++j) {
                const int m = n0 + (mt0 + mt) * 16 + j + hi * 8;
                out[(size_t)m * CDIM + c] = acc[mt][nt][j] + bval;
            }
        }
}

// ---------------------------------------------------------------------------
// Host launcher
// ---------------------------------------------------------------------------
extern "C" void kernel_launch(void* const* d_in, const int* in_sizes, int n_in,
                              void* d_out, int out_size, void* d_ws, size_t ws_size,
                              hipStream_t stream) {
    const float* x        = (const float*)d_in[0];
    const float* qkv_w    = (const float*)d_in[1];
    const float* qkv_b    = (const float*)d_in[2];
    const float* proj_w   = (const float*)d_in[3];
    const float* proj_b   = (const float*)d_in[4];
    const float* q_norm_w = (const float*)d_in[5];
    const float* k_norm_w = (const float*)d_in[6];
    float* out = (float*)d_out;

    char* ws = (char*)d_ws;
    const size_t XB_OFF    = 0;
    const size_t WQKV_OFF  = XB_OFF    + (size_t)NTOK * CDIM * 2;
    const size_t WPROJ_OFF = WQKV_OFF  + (size_t)3 * CDIM * CDIM * 2;
    const size_t QB_OFF    = WPROJ_OFF + (size_t)CDIM * CDIM * 2;
    const size_t KB_OFF    = QB_OFF    + (size_t)NTOK * CDIM * 2;
    const size_t VT_OFF    = KB_OFF    + (size_t)NTOK * CDIM * 2;
    const size_t AB_OFF    = VT_OFF    + (size_t)NTOK * CDIM * 2;
    const size_t NEEDED    = AB_OFF    + (size_t)NTOK * CDIM * 2;   // 72 MB
    if (ws_size < NEEDED) return;

    unsigned short* xb    = (unsigned short*)(ws + XB_OFF);
    unsigned short* wqkv  = (unsigned short*)(ws + WQKV_OFF);
    unsigned short* wproj = (unsigned short*)(ws + WPROJ_OFF);
    unsigned short* qbuf  = (unsigned short*)(ws + QB_OFF);
    unsigned short* kbuf  = (unsigned short*)(ws + KB_OFF);
    unsigned short* vtbuf = (unsigned short*)(ws + VT_OFF);
    unsigned short* abuf  = (unsigned short*)(ws + AB_OFF);

    const int SM_QKV  = 128 * KPAD * 2 + 256 * 4;      // A chunk (68KB) + red
    const int SM_ATTN = 16 * NTOK * 4 + 2 * 128 * 4;   // 128KB S + red
    const int SM_PROJ = 128 * KPAD * 2;                // 68KB
    hipFuncSetAttribute((const void*)qkv_gemm_fused_kernel,
                        hipFuncAttributeMaxDynamicSharedMemorySize, SM_QKV);
    hipFuncSetAttribute((const void*)attn_kernel,
                        hipFuncAttributeMaxDynamicSharedMemorySize, SM_ATTN);
    hipFuncSetAttribute((const void*)proj_gemm_kernel,
                        hipFuncAttributeMaxDynamicSharedMemorySize, SM_PROJ);

    cvt_f32_bf16_kernel<<<2048, 256, 0, stream>>>(x, xb, NTOK * CDIM);
    cvt_f32_bf16_kernel<<<4096, 256, 0, stream>>>(qkv_w, wqkv, 3 * CDIM * CDIM);
    cvt_f32_bf16_kernel<<<2048, 256, 0, stream>>>(proj_w, wproj, CDIM * CDIM);

    qkv_gemm_fused_kernel<<<dim3(NTOK / 128, 3 * NHEAD), 256, SM_QKV, stream>>>(
        xb, wqkv, qkv_b, q_norm_w, k_norm_w, qbuf, kbuf, vtbuf);

    attn_kernel<<<dim3(NTOK / 16, NHEAD), 128, SM_ATTN, stream>>>(
        qbuf, kbuf, vtbuf, abuf);

    proj_gemm_kernel<<<dim3(NTOK / 128, CDIM / 128), 256, SM_PROJ, stream>>>(
        abuf, wproj, proj_b, out);
}
